// FOL_32487132627309
// MI455X (gfx1250) — compile-verified, matrix operand order checked by reading, not run on hardware
//
#include <hip/hip_runtime.h>

// ---------------------------------------------------------------------------
// Problem constants (from the reference)
// ---------------------------------------------------------------------------
constexpr int Bc  = 512;   // batch
constexpr int Sc  = 10;    // segments
constexpr int Tc  = 10;    // decoder timesteps
constexpr int Ec  = 512;   // embed dim
constexpr int HBc = 512;   // box GRU hidden
constexpr int HDc = 512;   // dm GRU hidden
constexpr int Pc  = 512;   // predictor input dim
constexpr int PDc = 4;     // prediction dim
constexpr int Dc  = HBc + HDc; // 1024 decoder hidden

typedef __attribute__((ext_vector_type(16))) __bf16 v16bf;
typedef __attribute__((ext_vector_type(8)))  __bf16 bfx8;
typedef __attribute__((ext_vector_type(8)))  float  v8f;

union FragU { v16bf v; bfx8 h[2]; };

// Load a 16x32 bf16 WMMA A/B fragment (NT layout: row-major [16 rows, K]).
// ISA layout (16-bit A/B, wave32): lane&15 = row, lane>>4 selects K-half:
//   elements 0..7  = K = kb + (lane>>4)*8 .. +8
//   elements 8..15 = K = kb + 16 + (lane>>4)*8 .. +8
__device__ __forceinline__ v16bf load_frag(const __bf16* __restrict__ base,
                                           int ld, int row, int kb, int kh) {
    const __bf16* p = base + (size_t)row * ld + kb + kh * 8;
    FragU f;
    f.h[0] = *reinterpret_cast<const bfx8*>(p);
    f.h[1] = *reinterpret_cast<const bfx8*>(p + 16);
    return f.v;
}

__device__ __forceinline__ float sigmoidf_(float x) {
    return 1.0f / (1.0f + __expf(-x));
}

// ---------------------------------------------------------------------------
// Fully fused GRU cell step (PyTorch GRUCell semantics):
//   h' = GRU(x, h)  with  gi = x@Wih^T + bih, gh = h@Whh^T + bhh
// Each wave owns a 16(M) x 16(N) tile of the hidden state and accumulates all
// three gates for gi and gh via WMMA (6 fp32 accumulators), then applies the
// sigmoid/tanh gate math in-register and writes h' (fp32 + bf16 twin + opt
// extra strided copy for the segment_dec_h output).
// h is double-buffered by the host (read Hb/Hprev, write Hf/Hbf) so no block
// can overwrite state another block still reads.
// Grid: (H/16, B/128), block 256 threads = 8 waves.
// ---------------------------------------------------------------------------
template<bool HAS_EXTRA>
__global__ __launch_bounds__(256) void gru_gemm_fused(
    const __bf16* __restrict__ X,  int ldx,                     // [B,Kx]
    const __bf16* __restrict__ Wih, const float* __restrict__ bih, int Kx, // [3H,Kx]
    const __bf16* __restrict__ Hb, int ldh,                     // [B,*] bf16 h (col off folded)
    const __bf16* __restrict__ Whh, const float* __restrict__ bhh, int Kh, // [3H,Kh]
    const float* __restrict__ Hprev,                            // fp32 h (col off folded)
    float* __restrict__ Hf, __bf16* __restrict__ Hbf,           // h' out (col off folded)
    float* __restrict__ extra, long estride,
    int H)
{
    const int lane = threadIdx.x & 31;
    const int wave = threadIdx.x >> 5;
    const int m0   = blockIdx.y * 128 + wave * 16;
    const int n0   = blockIdx.x * 16;
    const int r    = lane & 15;
    const int kh   = lane >> 4;

    v8f ai[3], ah[3];
#pragma unroll
    for (int g = 0; g < 3; ++g) { ai[g] = 0.0f; ah[g] = 0.0f; }

    // gi = x @ Wih^T   (gate g lives in Wih rows [g*H + n0, +16))
    for (int kb = 0; kb < Kx; kb += 32) {
        v16bf a = load_frag(X, ldx, m0 + r, kb, kh);
#pragma unroll
        for (int g = 0; g < 3; ++g) {
            v16bf b = load_frag(Wih, Kx, g * H + n0 + r, kb, kh);
            ai[g] = __builtin_amdgcn_wmma_f32_16x16x32_bf16(false, a, false, b, (short)0, ai[g], false, false);
        }
    }
    // gh = h @ Whh^T
    for (int kb = 0; kb < Kh; kb += 32) {
        v16bf a = load_frag(Hb, ldh, m0 + r, kb, kh);
#pragma unroll
        for (int g = 0; g < 3; ++g) {
            v16bf b = load_frag(Whh, Kh, g * H + n0 + r, kb, kh);
            ah[g] = __builtin_amdgcn_wmma_f32_16x16x32_bf16(false, a, false, b, (short)0, ah[g], false, false);
        }
    }

    // C/D layout: lane&15 = column, (lane>>4)*8 + vgpr = row
    const int mb = m0 + kh * 8;
    const int j  = n0 + r;
    const float bir = bih[j], biz = bih[H + j], bin = bih[2 * H + j];
    const float bhr = bhh[j], bhz = bhh[H + j], bhn = bhh[2 * H + j];
#pragma unroll
    for (int rr = 0; rr < 8; ++rr) {
        const int row = mb + rr;
        const float rg = sigmoidf_(ai[0][rr] + bir + ah[0][rr] + bhr);
        const float zg = sigmoidf_(ai[1][rr] + biz + ah[1][rr] + bhz);
        const float ng = tanhf(ai[2][rr] + bin + rg * (ah[2][rr] + bhn));
        const float hp = Hprev[(size_t)row * ldh + j];
        const float hv = (1.0f - zg) * ng + zg * hp;
        Hf [(size_t)row * ldh + j] = hv;
        Hbf[(size_t)row * ldh + j] = (__bf16)hv;
        if (HAS_EXTRA) extra[(size_t)row * estride + j] = hv;
    }
}

// ---------------------------------------------------------------------------
// C[M,N] = act(A[M,K] @ B[N,K]^T + bias[N]); A,B bf16, C fp32 (+opt bf16 twin)
// Grid: (N/64, M/128), block 256 threads = 8 waves; each wave: 16(M) x 64(N).
// ---------------------------------------------------------------------------
template<int ACT, bool WRITE_BF16>
__global__ __launch_bounds__(256) void gemm_nt_bf16(
    const __bf16* __restrict__ A, int lda,
    const __bf16* __restrict__ Bm, int ldb,
    float* __restrict__ C, __bf16* __restrict__ Cbf, int ldc,
    const float* __restrict__ bias, int K)
{
    const int lane = threadIdx.x & 31;
    const int wave = threadIdx.x >> 5;
    const int m0   = blockIdx.y * 128 + wave * 16;
    const int n0   = blockIdx.x * 64;
    const int r    = lane & 15;
    const int kh   = lane >> 4;

    v8f acc[4];
#pragma unroll
    for (int t = 0; t < 4; ++t) acc[t] = 0.0f;

    for (int kb = 0; kb < K; kb += 32) {
        v16bf a  = load_frag(A,  lda, m0 + r,      kb, kh);
        v16bf b0 = load_frag(Bm, ldb, n0 +      r, kb, kh);
        v16bf b1 = load_frag(Bm, ldb, n0 + 16 + r, kb, kh);
        v16bf b2 = load_frag(Bm, ldb, n0 + 32 + r, kb, kh);
        v16bf b3 = load_frag(Bm, ldb, n0 + 48 + r, kb, kh);
        acc[0] = __builtin_amdgcn_wmma_f32_16x16x32_bf16(false, a, false, b0, (short)0, acc[0], false, false);
        acc[1] = __builtin_amdgcn_wmma_f32_16x16x32_bf16(false, a, false, b1, (short)0, acc[1], false, false);
        acc[2] = __builtin_amdgcn_wmma_f32_16x16x32_bf16(false, a, false, b2, (short)0, acc[2], false, false);
        acc[3] = __builtin_amdgcn_wmma_f32_16x16x32_bf16(false, a, false, b3, (short)0, acc[3], false, false);
    }

    const int mb = m0 + kh * 8;
#pragma unroll
    for (int t = 0; t < 4; ++t) {
        const int n  = n0 + t * 16 + r;
        const float bv = bias ? bias[n] : 0.0f;
#pragma unroll
        for (int rr = 0; rr < 8; ++rr) {
            float v = acc[t][rr] + bv;
            if (ACT == 1) v = fmaxf(v, 0.0f);
            const size_t off = (size_t)(mb + rr) * ldc + n;
            C[off] = v;
            if (WRITE_BF16) Cbf[off] = (__bf16)v;
        }
    }
}

// out[b, :] = tanh(h[b, :] @ W[D,PD] + bias)  -> strided slice of fol output
__global__ void h2p_tanh(const float* __restrict__ h, const float* __restrict__ W,
                         const float* __restrict__ bias,
                         float* __restrict__ out, long ostride, int K)
{
    int i = blockIdx.x * blockDim.x + threadIdx.x;
    if (i >= Bc * PDc) return;
    int b = i / PDc, p = i - b * PDc;
    const float* hr = h + (size_t)b * K;
    float acc = bias[p];
    for (int k = 0; k < K; ++k) acc += hr[k] * W[(size_t)k * PDc + p];
    out[(size_t)b * ostride + p] = tanhf(acc);
}

// emb = relu(x[rows,4] @ W[4,E] + b[E]) -> bf16 (consumed only as GEMM A)
__global__ void embed_relu_bf16(const float* __restrict__ x, const float* __restrict__ W,
                                const float* __restrict__ bias,
                                __bf16* __restrict__ out, int rows)
{
    int i = blockIdx.x * blockDim.x + threadIdx.x;
    if (i >= rows * Ec) return;
    int rI = i / Ec, e = i - rI * Ec;
    const float* xr = x + (size_t)rI * 4;
    float acc = bias[e] + xr[0] * W[e] + xr[1] * W[Ec + e]
                        + xr[2] * W[2 * Ec + e] + xr[3] * W[3 * Ec + e];
    out[i] = (__bf16)fmaxf(acc, 0.0f);
}

__global__ void cvt_bf16(const float* __restrict__ in, __bf16* __restrict__ out, int n)
{
    int i = blockIdx.x * blockDim.x + threadIdx.x;
    if (i < n) out[i] = (__bf16)in[i];
}

// in [R,C] -> out [C,R] bf16
__global__ void transpose_cvt_bf16(const float* __restrict__ in, __bf16* __restrict__ out,
                                   int R, int C)
{
    int i = blockIdx.x * blockDim.x + threadIdx.x;
    if (i >= R * C) return;
    int c = i / R, r = i - c * R;
    out[(size_t)c * R + r] = (__bf16)in[(size_t)r * C + c];
}

__global__ void zero_h(float* __restrict__ hf, __bf16* __restrict__ hb, int n)
{
    int i = blockIdx.x * blockDim.x + threadIdx.x;
    if (i < n) { hf[i] = 0.0f; hb[i] = (__bf16)0.0f; }
}

// h_dec <- h_enc (fp32 + bf16), pi <- 0
__global__ void init_dec(const float* __restrict__ hef, const __bf16* __restrict__ heb,
                         float* __restrict__ hdf, __bf16* __restrict__ hdb,
                         float* __restrict__ pif, __bf16* __restrict__ pib)
{
    int i = blockIdx.x * blockDim.x + threadIdx.x;
    if (i >= Bc * Dc) return;
    hdf[i] = hef[i];
    hdb[i] = heb[i];
    int b = i / Dc, j = i - b * Dc;
    if (j < Pc) {
        pif[(size_t)b * Pc + j] = 0.0f;
        pib[(size_t)b * Pc + j] = (__bf16)0.0f;
    }
}

// ---------------------------------------------------------------------------
// Host orchestration
// ---------------------------------------------------------------------------
extern "C" void kernel_launch(void* const* d_in, const int* in_sizes, int n_in,
                              void* d_out, int out_size, void* d_ws, size_t ws_size,
                              hipStream_t stream)
{
    const float* box     = (const float*)d_in[0];
    const float* dm      = (const float*)d_in[1];
    const float* Wbe     = (const float*)d_in[2];
    const float* bbe     = (const float*)d_in[3];
    const float* Wde     = (const float*)d_in[4];
    const float* bde     = (const float*)d_in[5];
    const float* Wih_box = (const float*)d_in[6];
    const float* Whh_box = (const float*)d_in[7];
    const float* bih_box = (const float*)d_in[8];
    const float* bhh_box = (const float*)d_in[9];
    const float* Wih_dm  = (const float*)d_in[10];
    const float* Whh_dm  = (const float*)d_in[11];
    const float* bih_dm  = (const float*)d_in[12];
    const float* bhh_dm  = (const float*)d_in[13];
    const float* Wih_dec = (const float*)d_in[14];
    const float* Whh_dec = (const float*)d_in[15];
    const float* bih_dec = (const float*)d_in[16];
    const float* bhh_dec = (const float*)d_in[17];
    const float* W_h2pi  = (const float*)d_in[18];
    const float* b_h2pi  = (const float*)d_in[19];
    const float* W_h2p   = (const float*)d_in[20];
    const float* b_h2p   = (const float*)d_in[21];

    float* fol  = (float*)d_out;                          // [B,S,T,PD]
    float* dech = fol + (size_t)Bc * Sc * Tc * PDc;       // [B,S,T,D]

    // Workspace bump allocator
    char* wp = (char*)d_ws;
    auto alloc = [&](size_t bytes) -> char* {
        char* p = wp; wp += (bytes + 255) & ~(size_t)255; return p;
    };
    __bf16* ebox    = (__bf16*)alloc((size_t)Bc * Sc * Ec * 2);
    __bf16* edm     = (__bf16*)alloc((size_t)Bc * Sc * Ec * 2);
    __bf16* wihbox  = (__bf16*)alloc((size_t)3 * HBc * Ec * 2);
    __bf16* whhbox  = (__bf16*)alloc((size_t)3 * HBc * HBc * 2);
    __bf16* wihdm   = (__bf16*)alloc((size_t)3 * HDc * Ec * 2);
    __bf16* whhdm   = (__bf16*)alloc((size_t)3 * HDc * HDc * 2);
    __bf16* wihdec  = (__bf16*)alloc((size_t)3 * Dc * Pc * 2);
    __bf16* whhdec  = (__bf16*)alloc((size_t)3 * Dc * Dc * 2);
    __bf16* wh2piT  = (__bf16*)alloc((size_t)Pc * Dc * 2);   // [P,D]
    float*  henc_f[2]; __bf16* henc_b[2];
    float*  hdec_f[2]; __bf16* hdec_b[2];
    for (int i = 0; i < 2; ++i) {
        henc_f[i] = (float*) alloc((size_t)Bc * Dc * 4);
        henc_b[i] = (__bf16*)alloc((size_t)Bc * Dc * 2);
        hdec_f[i] = (float*) alloc((size_t)Bc * Dc * 4);
        hdec_b[i] = (__bf16*)alloc((size_t)Bc * Dc * 2);
    }
    float*  pi_f = (float*) alloc((size_t)Bc * Pc * 4);
    __bf16* pi_b = (__bf16*)alloc((size_t)Bc * Pc * 2);
    (void)ws_size; (void)in_sizes; (void)n_in; (void)out_size;

    auto blocks = [](size_t n) { return (unsigned)((n + 255) / 256); };
    auto cvt = [&](const float* in, __bf16* out, size_t n) {
        cvt_bf16<<<blocks(n), 256, 0, stream>>>(in, out, (int)n);
    };
    // Fused GRU step: h' = GRU(x, h)
    auto gru = [&](const __bf16* X, int ldx,
                   const __bf16* Wih, const float* bih, int Kx,
                   const __bf16* Hb, const __bf16* Whh, const float* bhh, int Kh,
                   const float* Hprev, float* Hf, __bf16* Hbf,
                   float* extra, long estride, int H) {
        dim3 g(H / 16, Bc / 128);
        if (extra)
            gru_gemm_fused<true ><<<g, 256, 0, stream>>>(X, ldx, Wih, bih, Kx, Hb, Dc,
                                                         Whh, bhh, Kh, Hprev, Hf, Hbf,
                                                         extra, estride, H);
        else
            gru_gemm_fused<false><<<g, 256, 0, stream>>>(X, ldx, Wih, bih, Kx, Hb, Dc,
                                                         Whh, bhh, Kh, Hprev, Hf, Hbf,
                                                         nullptr, 0, H);
    };

    // ---- one-time (per launch) weight conversion ----
    cvt(Wih_box, wihbox, (size_t)3 * HBc * Ec);
    cvt(Whh_box, whhbox, (size_t)3 * HBc * HBc);
    cvt(Wih_dm,  wihdm,  (size_t)3 * HDc * Ec);
    cvt(Whh_dm,  whhdm,  (size_t)3 * HDc * HDc);
    cvt(Wih_dec, wihdec, (size_t)3 * Dc * Pc);
    cvt(Whh_dec, whhdec, (size_t)3 * Dc * Dc);
    transpose_cvt_bf16<<<blocks((size_t)Dc * Pc), 256, 0, stream>>>(W_h2pi, wh2piT, Dc, Pc);

    // ---- embeddings (K=4, VALU) ----
    embed_relu_bf16<<<blocks((size_t)Bc * Sc * Ec), 256, 0, stream>>>(box, Wbe, bbe, ebox, Bc * Sc);
    embed_relu_bf16<<<blocks((size_t)Bc * Sc * Ec), 256, 0, stream>>>(dm,  Wde, bde, edm,  Bc * Sc);

    // ---- encoder hidden = 0 (buffer 0) ----
    zero_h<<<blocks((size_t)Bc * Dc), 256, 0, stream>>>(henc_f[0], henc_b[0], Bc * Dc);

    int ec = 0;
    for (int s = 0; s < Sc; ++s) {
        const int en = 1 - ec;
        // box GRU: hidden lives in columns [0, HB) of the concat buffer
        gru(ebox + (size_t)s * Ec, Sc * Ec, wihbox, bih_box, Ec,
            henc_b[ec], whhbox, bhh_box, HBc,
            henc_f[ec], henc_f[en], henc_b[en], nullptr, 0, HBc);
        // dm GRU: hidden lives in columns [HB, D)
        gru(edm + (size_t)s * Ec, Sc * Ec, wihdm, bih_dm, Ec,
            henc_b[ec] + HBc, whhdm, bhh_dm, HDc,
            henc_f[ec] + HBc, henc_f[en] + HBc, henc_b[en] + HBc, nullptr, 0, HDc);
        ec = en;

        // decoder init: h = concat(box_h, dm_h) (already contiguous), pi = 0
        init_dec<<<blocks((size_t)Bc * Dc), 256, 0, stream>>>(
            henc_f[ec], henc_b[ec], hdec_f[0], hdec_b[0], pi_f, pi_b);

        int dc_ = 0;
        for (int t = 0; t < Tc; ++t) {
            const int dn = 1 - dc_;
            // h = GRU(pi, h); streams h straight into segment_dec_h slice
            gru(pi_b, Pc, wihdec, bih_dec, Pc,
                hdec_b[dc_], whhdec, bhh_dec, Dc,
                hdec_f[dc_], hdec_f[dn], hdec_b[dn],
                dech + (size_t)(s * Tc + t) * Dc, (long)Sc * Tc * Dc, Dc);
            dc_ = dn;
            // pi = relu(h @ W_h2pi + b)  (fp32 + bf16 twin)
            {
                dim3 g(Pc / 64, Bc / 128);
                gemm_nt_bf16<1, true><<<g, 256, 0, stream>>>(
                    hdec_b[dc_], Dc, wh2piT, Dc, pi_f, pi_b, Pc, b_h2pi, Dc);
            }
            // out = tanh(h @ W_h2p + b)  (N=4, VALU)
            h2p_tanh<<<blocks((size_t)Bc * PDc), 256, 0, stream>>>(
                hdec_f[dc_], W_h2p, b_h2p, fol + (size_t)(s * Tc + t) * PDc,
                (long)Sc * Tc * PDc, Dc);
        }
    }
}
